// DSNN_28114855920459
// MI455X (gfx1250) — matrix operand order repaired and checked by reading, not compiled, not run on hardware
//
#include <hip/hip_runtime.h>
#include <hip/hip_bf16.h>

// ---------------------------------------------------------------------------
// DSNN forward, restructured for MI455X (gfx1250):
//   h0 = [x, 0.5] @ w0                (one small WMMA GEMM, time-invariant)
//   scan0: elementwise LIF over T     -> mem_rec0, spk_rec0, spk0(f16)
//   H1 = spk0_all_t @ w1              (one big WMMA GEMM, [T*B,512]x[512,512])
//   scan1: elementwise LIF over T     -> mem_rec1, spk_rec1, spk1(f16)
//   H2 = spk1_all_t @ w2(padded N=16) (WMMA GEMM, [T*B,512]x[512,16])
//   scan2: elementwise LIF over T     -> mem_rec2, spk_rec2, final membrane
// GEMMs are register-tiled (MT x NT 16x16 WMMA tiles per wave) and software
// pipelined (double-buffered fragments) so WMMAs overlap in-flight loads.
// ---------------------------------------------------------------------------

typedef __attribute__((ext_vector_type(16))) _Float16 v16h;
typedef __attribute__((ext_vector_type(8)))  _Float16 v8h;
typedef __attribute__((ext_vector_type(8)))  float    v8f;

#define SNN_B     2048
#define SNN_DIN   128
#define SNN_H     512
#define SNN_NO    6
#define SNN_T     64
#define SNN_TB    (SNN_T * SNN_B)       // 131072
#define SNN_BH    ((size_t)SNN_B * SNN_H)
#define ALPHA_F   0.95f
#define BETA_F    0.9f

// ---- output (d_out, float) offsets, in elements, return order of reference ----
#define OUT_FINAL  ((size_t)0)                                   // [2048,6]
#define OUT_MEM0   (OUT_FINAL + (size_t)SNN_B * SNN_NO)          // [64,2048,512]
#define OUT_MEM1   (OUT_MEM0 + (size_t)SNN_T * SNN_B * SNN_H)
#define OUT_MEM2   (OUT_MEM1 + (size_t)SNN_T * SNN_B * SNN_H)    // [64,2048,6]
#define OUT_SPK0   (OUT_MEM2 + (size_t)SNN_T * SNN_B * SNN_NO)
#define OUT_SPK1   (OUT_SPK0 + (size_t)SNN_T * SNN_B * SNN_H)
#define OUT_SPK2   (OUT_SPK1 + (size_t)SNN_T * SNN_B * SNN_H)

// ---- workspace byte offsets (all 256B aligned) ----
#define WS_XF16    ((size_t)0)                         // 2048*128 f16      = 512KB
#define WS_W0T     (WS_XF16  + (size_t)524288)         // 512*128 f16       = 128KB
#define WS_W0B     (WS_W0T   + (size_t)131072)         // 512 f32           = 2KB
#define WS_W1T     (WS_W0B   + (size_t)2048)           // 512*512 f16       = 512KB
#define WS_W2T     (WS_W1T   + (size_t)524288)         // 16*512 f16        = 16KB
#define WS_H0      (WS_W2T   + (size_t)16384)          // 2048*512 f32      = 4MB
#define WS_SPK0H   (WS_H0    + (size_t)4194304)        // 131072*512 f16    = 128MB
#define WS_H1      (WS_SPK0H + (size_t)134217728)      // 131072*512 f32    = 256MB
#define WS_SPK1H   (WS_H1    + (size_t)268435456)      // 131072*512 f16    = 128MB
#define WS_H2      (WS_SPK1H + (size_t)134217728)      // 131072*16 f32     = 8MB

// ===========================================================================
// Prep kernels: f32 -> f16 conversion and weight transposes (tiny, run once).
// ===========================================================================
__global__ void snn_cvt_inputs(const float* __restrict__ in, _Float16* __restrict__ out, int n) {
    int i = blockIdx.x * blockDim.x + threadIdx.x;
    if (i < n) out[i] = (_Float16)in[i];
}

// w0 is [129,512] row-major; build w0t[n][k] = w0[k][n] (k<128), and bias row.
__global__ void snn_w0_prep(const float* __restrict__ w0, _Float16* __restrict__ w0t,
                            float* __restrict__ w0b) {
    int i = blockIdx.x * blockDim.x + threadIdx.x;           // 512*128 threads
    if (i < SNN_H * SNN_DIN) {
        int n = i / SNN_DIN, k = i % SNN_DIN;
        w0t[i] = (_Float16)w0[(size_t)k * SNN_H + n];
    }
    if (i < SNN_H) w0b[i] = w0[(size_t)SNN_DIN * SNN_H + i]; // row 128 (bias)
}

__global__ void snn_w1_prep(const float* __restrict__ w1, _Float16* __restrict__ w1t) {
    int i = blockIdx.x * blockDim.x + threadIdx.x;           // 512*512 threads
    if (i < SNN_H * SNN_H) {
        int n = i / SNN_H, k = i % SNN_H;
        w1t[i] = (_Float16)w1[(size_t)k * SNN_H + n];
    }
}

// w2 is [512,6]; build padded transpose w2t[16][512], columns 6..15 zero.
__global__ void snn_w2_prep(const float* __restrict__ w2, _Float16* __restrict__ w2t) {
    int i = blockIdx.x * blockDim.x + threadIdx.x;           // 16*512 threads
    if (i < 16 * SNN_H) {
        int n = i / SNN_H, k = i % SNN_H;
        w2t[i] = (n < SNN_NO) ? (_Float16)w2[(size_t)k * SNN_NO + n] : (_Float16)0.0f;
    }
}

// ===========================================================================
// WMMA GEMM: C[M,N](f32) = A[M,K](f16, row-major) * Bt[N,K](f16, i.e. B^T)
//            (+ bias_scale * bias[n] if bias != nullptr)
// Each wave owns an (MT*16) x (NT*16) C macro-tile; fragments double-buffered
// so the K-step's WMMAs run against resident registers while the next step's
// global_load_b128 clauses are in flight. 8 waves/block. K multiple of 32.
// A/B fragments: contiguous K (lane half h holds K 16h..16h+15); C layout per CDNA5 ISA 7.12.2 (wave32).
// ===========================================================================
template <int MT, int NT>
__global__ void snn_gemm_wmma(const _Float16* __restrict__ A, int lda,
                              const _Float16* __restrict__ Bt, int ldb,
                              float* __restrict__ C, int ldc,
                              int M, int N, int K,
                              const float* __restrict__ bias, float bias_scale) {
    const int ntg   = N / (16 * NT);
    const int total = (M / (16 * MT)) * ntg;
    const int tile  = blockIdx.x * (blockDim.x >> 5) + (threadIdx.x >> 5);
    if (tile >= total) return;                       // uniform per wave: EXEC stays full
    const int mi = tile / ntg;
    const int ni = tile - mi * ntg;
    const int lane = threadIdx.x & 31;
    const int l16  = lane >> 4;                      // 0: lanes 0-15, 1: lanes 16-31

    // A frags: row = mi*MT*16 + mt*16 + (lane&15); 16 consecutive K at +16*l16.
    const _Float16* Ap[MT];
#pragma unroll
    for (int mt = 0; mt < MT; ++mt)
        Ap[mt] = A + (size_t)(mi * MT * 16 + mt * 16 + (lane & 15)) * lda + (l16 << 4);
    // B frags: col = ni*NT*16 + nt*16 + (lane&15); 16 consecutive K at +16*l16.
    const _Float16* Bp[NT];
#pragma unroll
    for (int nt = 0; nt < NT; ++nt)
        Bp[nt] = Bt + (size_t)(ni * NT * 16 + nt * 16 + (lane & 15)) * ldb + (l16 << 4);

    auto load_frags = [&](int kb, v16h (&a)[MT], v16h (&b)[NT]) {
#pragma unroll
        for (int mt = 0; mt < MT; ++mt) {
            v8h a0 = *(const v8h*)(Ap[mt] + kb);        // K kb..+7 (lanes0-15) / +16..23
            v8h a1 = *(const v8h*)(Ap[mt] + kb + 8);    // K +8..15 / +24..31
            a[mt] = __builtin_shufflevector(a0, a1, 0, 1, 2, 3, 4, 5, 6, 7,
                                                    8, 9, 10, 11, 12, 13, 14, 15);
        }
#pragma unroll
        for (int nt = 0; nt < NT; ++nt)
            b[nt] = *(const v16h*)(Bp[nt] + kb);        // 16 K values of one column
    };

    v8f acc[MT][NT] = {};
    v16h a_cur[MT], b_cur[NT];
    load_frags(0, a_cur, b_cur);

    int kb = 0;
    for (; kb + 32 < K; kb += 32) {
        v16h a_nxt[MT], b_nxt[NT];
        load_frags(kb + 32, a_nxt, b_nxt);              // in-flight during WMMAs below
        __builtin_prefetch(Ap[0] + kb + 256, 0, 3);     // pull A stream into near caches
#pragma unroll
        for (int mt = 0; mt < MT; ++mt)
#pragma unroll
            for (int nt = 0; nt < NT; ++nt)
                acc[mt][nt] = __builtin_amdgcn_wmma_f32_16x16x32_f16(
                    false, a_cur[mt], false, b_cur[nt], (short)0, acc[mt][nt], false, false);
#pragma unroll
        for (int mt = 0; mt < MT; ++mt) a_cur[mt] = a_nxt[mt];
#pragma unroll
        for (int nt = 0; nt < NT; ++nt) b_cur[nt] = b_nxt[nt];
    }
    // epilogue K-step
#pragma unroll
    for (int mt = 0; mt < MT; ++mt)
#pragma unroll
        for (int nt = 0; nt < NT; ++nt)
            acc[mt][nt] = __builtin_amdgcn_wmma_f32_16x16x32_f16(
                false, a_cur[mt], false, b_cur[nt], (short)0, acc[mt][nt], false, false);

    if (bias) {
#pragma unroll
        for (int nt = 0; nt < NT; ++nt) {
            const float bv = bias_scale * bias[ni * NT * 16 + nt * 16 + (lane & 15)];
#pragma unroll
            for (int mt = 0; mt < MT; ++mt)
#pragma unroll
                for (int r = 0; r < 8; ++r) acc[mt][nt][r] += bv;
        }
    }

    // C layout: lane 0-15 -> N=lane, VGPR r -> M=r; lanes 16-31 -> M=r+8.
#pragma unroll
    for (int mt = 0; mt < MT; ++mt) {
#pragma unroll
        for (int nt = 0; nt < NT; ++nt) {
            float* Cp = C + (size_t)(mi * MT * 16 + mt * 16 + (l16 << 3)) * ldc
                          + (ni * NT * 16 + nt * 16 + (lane & 15));
#pragma unroll
            for (int r = 0; r < 8; ++r) Cp[(size_t)r * ldc] = acc[mt][nt][r];
        }
    }
}

// ===========================================================================
// Elementwise LIF scans (recurrence only dimension is t).
// Output records are write-once -> non-temporal stores keep L2 for operands.
// ===========================================================================
// Layer 0: syn = h0 every step (no alpha accumulation at l==0).
__global__ void snn_scan0(const float* __restrict__ h0, float* __restrict__ out,
                          _Float16* __restrict__ spk0h) {
    size_t idx = (size_t)blockIdx.x * blockDim.x + threadIdx.x;
    if (idx >= SNN_BH) return;
    const float h = h0[idx];
    float mem = 0.0f;
#pragma unroll 4
    for (int t = 0; t < SNN_T; ++t) {
        mem = BETA_F * mem + h;
        const bool fire = (mem - 1.0f) > 0.0f;
        const float s = fire ? 1.0f : 0.0f;
        mem = fire ? 0.0f : mem;                     // hidden layer: reset
        const size_t o = (size_t)t * SNN_BH + idx;
        __builtin_nontemporal_store(mem, &out[OUT_MEM0 + o]);
        __builtin_nontemporal_store(s,   &out[OUT_SPK0 + o]);
        spk0h[o] = (_Float16)s;
    }
}

// Layer 1: syn = alpha*syn + h1[t].
__global__ void snn_scan1(const float* __restrict__ H1, float* __restrict__ out,
                          _Float16* __restrict__ spk1h) {
    size_t idx = (size_t)blockIdx.x * blockDim.x + threadIdx.x;
    if (idx >= SNN_BH) return;
    float syn = 0.0f, mem = 0.0f;
#pragma unroll 2
    for (int t = 0; t < SNN_T; ++t) {
        const size_t o = (size_t)t * SNN_BH + idx;
        syn = ALPHA_F * syn + __builtin_nontemporal_load(&H1[o]);  // read-once stream
        mem = BETA_F * mem + syn;
        const bool fire = (mem - 1.0f) > 0.0f;
        const float s = fire ? 1.0f : 0.0f;
        mem = fire ? 0.0f : mem;                     // hidden layer: reset
        __builtin_nontemporal_store(mem, &out[OUT_MEM1 + o]);
        __builtin_nontemporal_store(s,   &out[OUT_SPK1 + o]);
        spk1h[o] = (_Float16)s;
    }
}

// Layer 2 (output): no reset; also emits final membrane (reference output 0).
__global__ void snn_scan2(const float* __restrict__ H2, float* __restrict__ out) {
    int idx = blockIdx.x * blockDim.x + threadIdx.x;
    if (idx >= SNN_B * SNN_NO) return;
    const int b = idx / SNN_NO, j = idx - b * SNN_NO;
    float syn = 0.0f, mem = 0.0f;
#pragma unroll 4
    for (int t = 0; t < SNN_T; ++t) {
        const float h = H2[(size_t)t * SNN_B * 16 + (size_t)b * 16 + j];  // ldc=16 padded
        syn = ALPHA_F * syn + h;
        mem = BETA_F * mem + syn;
        const float s = ((mem - 1.0f) > 0.0f) ? 1.0f : 0.0f;
        const size_t o = (size_t)t * SNN_B * SNN_NO + idx;
        __builtin_nontemporal_store(mem, &out[OUT_MEM2 + o]);  // no reset on last layer
        __builtin_nontemporal_store(s,   &out[OUT_SPK2 + o]);
    }
    out[OUT_FINAL + idx] = mem;                      // mems_final[-1]
}

// ===========================================================================
extern "C" void kernel_launch(void* const* d_in, const int* in_sizes, int n_in,
                              void* d_out, int out_size, void* d_ws, size_t ws_size,
                              hipStream_t stream) {
    (void)in_sizes; (void)n_in; (void)out_size; (void)ws_size;
    const float* inputs = (const float*)d_in[0];   // [2048,128]
    const float* w0     = (const float*)d_in[1];   // [129,512]
    const float* w1     = (const float*)d_in[2];   // [512,512]
    const float* w2     = (const float*)d_in[3];   // [512,6]
    float* out = (float*)d_out;

    char* ws = (char*)d_ws;
    _Float16* xf16  = (_Float16*)(ws + WS_XF16);
    _Float16* w0t   = (_Float16*)(ws + WS_W0T);
    float*    w0b   = (float*)   (ws + WS_W0B);
    _Float16* w1t   = (_Float16*)(ws + WS_W1T);
    _Float16* w2t   = (_Float16*)(ws + WS_W2T);
    float*    h0    = (float*)   (ws + WS_H0);
    _Float16* spk0h = (_Float16*)(ws + WS_SPK0H);
    float*    H1    = (float*)   (ws + WS_H1);
    _Float16* spk1h = (_Float16*)(ws + WS_SPK1H);
    float*    H2    = (float*)   (ws + WS_H2);

    const int TPB = 256;   // 8 wave32 waves per block

    // --- prep: conversions + transposes ---
    snn_cvt_inputs<<<(SNN_B * SNN_DIN + TPB - 1) / TPB, TPB, 0, stream>>>(
        inputs, xf16, SNN_B * SNN_DIN);
    snn_w0_prep<<<(SNN_H * SNN_DIN + TPB - 1) / TPB, TPB, 0, stream>>>(w0, w0t, w0b);
    snn_w1_prep<<<(SNN_H * SNN_H + TPB - 1) / TPB, TPB, 0, stream>>>(w1, w1t);
    snn_w2_prep<<<(16 * SNN_H + TPB - 1) / TPB, TPB, 0, stream>>>(w2, w2t);

    // --- h0 = x @ w0 + 0.5*bias_row : M=2048, N=512, K=128 (16x64 per wave) ---
    {
        int tiles = (SNN_B / 16) * (SNN_H / 64);
        snn_gemm_wmma<1, 4><<<(tiles + 7) / 8, TPB, 0, stream>>>(
            xf16, SNN_DIN, w0t, SNN_DIN, h0, SNN_H,
            SNN_B, SNN_H, SNN_DIN, w0b, 0.5f);
    }

    // --- layer-0 scan: produces spk0 for all T at once ---
    snn_scan0<<<(unsigned)((SNN_BH + TPB - 1) / TPB), TPB, 0, stream>>>(h0, out, spk0h);

    // --- H1 = spk0_all @ w1 : M=131072, N=512, K=512 (16x64 per wave) ---
    {
        int tiles = (SNN_TB / 16) * (SNN_H / 64);
        snn_gemm_wmma<1, 4><<<(tiles + 7) / 8, TPB, 0, stream>>>(
            spk0h, SNN_H, w1t, SNN_H, H1, SNN_H,
            SNN_TB, SNN_H, SNN_H, nullptr, 0.0f);
    }

    // --- layer-1 scan ---
    snn_scan1<<<(unsigned)((SNN_BH + TPB - 1) / TPB), TPB, 0, stream>>>(H1, out, spk1h);

    // --- H2 = spk1_all @ w2 : M=131072, N=16(padded), K=512 (64x16 per wave) ---
    {
        int tiles = (SNN_TB / 64) * (16 / 16);
        snn_gemm_wmma<4, 1><<<(tiles + 7) / 8, TPB, 0, stream>>>(
            spk1h, SNN_H, w2t, SNN_H, H2, 16,
            SNN_TB, 16, SNN_H, nullptr, 0.0f);
    }

    // --- layer-2 scan + final membrane ---
    snn_scan2<<<(SNN_B * SNN_NO + TPB - 1) / TPB, TPB, 0, stream>>>(H2, out);
}
